// RealStateSpaceModel_73675868996095
// MI455X (gfx1250) — compile-verified, matrix operand order checked by reading, not run on hardware
//
#include <hip/hip_runtime.h>
#include <hip/hip_bf16.h>
#include <stdint.h>

// ---------------------------------------------------------------------------
// Problem constants: batch=8, L=4096, D=256, 128 blocks of 2x2 rotations
// ---------------------------------------------------------------------------
#define BATCH 8
#define SEQ   4096
#define DIM   256
#define NBLK  128
#define NCH   64        // chunks for parallel scan
#define CHL   64        // chunk length (NCH*CHL == SEQ)
#define MTOT  (BATCH*SEQ)   // 32768 GEMM rows

typedef __attribute__((ext_vector_type(16))) __bf16 v16bf;
typedef __attribute__((ext_vector_type(2)))  __bf16 v2bf;
typedef __attribute__((ext_vector_type(8)))  float  v8f;

union FragB16 { uint32_t u[8]; v16bf v; };

// Native RNE fp32 -> bf16 (clang fptrunc -> v_cvt_pk_bf16_f32 on gfx1250)
__device__ __forceinline__ uint32_t pack_bf2(float lo, float hi) {
    v2bf t;
    t[0] = (__bf16)lo;
    t[1] = (__bf16)hi;
    return __builtin_bit_cast(uint32_t, t);
}
__device__ __forceinline__ uint16_t f2bf(float f) {
    __bf16 b = (__bf16)f;
    return __builtin_bit_cast(uint16_t, b);
}

// K offset of dword j within a 16-bit 16x32 WMMA fragment (ISA 7.12.2)
__device__ __forceinline__ int frag_k(int j, int half) {
    return ((j < 4) ? 2 * j : 16 + 2 * (j - 4)) + 8 * half;
}

// ---------------------------------------------------------------------------
// 0) Convert B, C (fp32 [k][n]) to bf16 transposed ([n][k]) in workspace
// ---------------------------------------------------------------------------
__global__ void k_convert_bc(const float* __restrict__ B, const float* __restrict__ C,
                             uint16_t* __restrict__ Bt, uint16_t* __restrict__ Ct) {
    int i = blockIdx.x * blockDim.x + threadIdx.x;   // 0..65535
    int k = i >> 8;
    int n = i & 255;
    Bt[n * DIM + k] = f2bf(B[k * DIM + n]);
    Ct[n * DIM + k] = f2bf(C[k * DIM + n]);
}

// ---------------------------------------------------------------------------
// 1) GEMM: D(MxN,fp32) = A(MxK) * W(KxN given as bf16 Wt[n][k])
//    A operand either fp32 (converted in-register) or bf16 (direct loads).
//    wave  -> 32 rows x 64 cols (2 A frags x 4 B frags = 8 WMMA / K-step)
//    block -> 8 waves = 256 rows x 64 cols;  grid = (M/256, N/64)
// ---------------------------------------------------------------------------
template <bool A_IS_BF16>
__global__ void __launch_bounds__(256)
k_gemm(const void* __restrict__ Aop, const uint16_t* __restrict__ Wt,
       float* __restrict__ D) {
    const int KD = DIM, N = DIM;
    const int lane = threadIdx.x & 31;
    const int wave = threadIdx.x >> 5;
    const int m0   = (blockIdx.x * 8 + wave) * 32;   // 32 rows per wave
    const int n0   = blockIdx.y * 64;
    const int l15  = lane & 15;
    const int half = lane >> 4;

    v8f acc[8];
    #pragma unroll
    for (int i = 0; i < 8; ++i) acc[i] = (v8f){};

    const float*    arowF0 = (const float*)Aop    + (size_t)(m0 +  0 + l15) * KD;
    const float*    arowF1 = (const float*)Aop    + (size_t)(m0 + 16 + l15) * KD;
    const uint16_t* arowH0 = (const uint16_t*)Aop + (size_t)(m0 +  0 + l15) * KD;
    const uint16_t* arowH1 = (const uint16_t*)Aop + (size_t)(m0 + 16 + l15) * KD;
    const uint16_t* brow[4];
    #pragma unroll
    for (int t = 0; t < 4; ++t)
        brow[t] = Wt + (size_t)(n0 + 16 * t + l15) * KD;

    for (int k0 = 0; k0 < KD; k0 += 32) {
        FragB16 a0, a1, b[4];
        #pragma unroll
        for (int j = 0; j < 8; ++j) {
            int kk = k0 + frag_k(j, half);
            if (A_IS_BF16) {
                a0.u[j] = *(const uint32_t*)(arowH0 + kk);
                a1.u[j] = *(const uint32_t*)(arowH1 + kk);
            } else {
                float2 f0 = *(const float2*)(arowF0 + kk);
                float2 f1 = *(const float2*)(arowF1 + kk);
                a0.u[j] = pack_bf2(f0.x, f0.y);
                a1.u[j] = pack_bf2(f1.x, f1.y);
            }
            #pragma unroll
            for (int t = 0; t < 4; ++t)
                b[t].u[j] = *(const uint32_t*)(brow[t] + kk);
        }
        #pragma unroll
        for (int t = 0; t < 4; ++t) {
            acc[t]     = __builtin_amdgcn_wmma_f32_16x16x32_bf16(false, a0.v, false, b[t].v,
                                                                 (short)0, acc[t],     false, false);
            acc[t + 4] = __builtin_amdgcn_wmma_f32_16x16x32_bf16(false, a1.v, false, b[t].v,
                                                                 (short)0, acc[t + 4], false, false);
        }
    }

    #pragma unroll
    for (int r = 0; r < 8; ++r) {
        size_t row0 = (size_t)(m0 +  0 + r + 8 * half) * N + n0 + l15;
        size_t row1 = (size_t)(m0 + 16 + r + 8 * half) * N + n0 + l15;
        #pragma unroll
        for (int t = 0; t < 4; ++t) {
            D[row0 + 16 * t] = acc[t][r];
            D[row1 + 16 * t] = acc[t + 4][r];
        }
    }
}

// ---------------------------------------------------------------------------
// 2a) Read-only local scan per (b, k, chunk): chunk final with zero carry
// ---------------------------------------------------------------------------
__global__ void __launch_bounds__(256)
k_scan_local(const float* __restrict__ x, const float* __restrict__ A,
             float* __restrict__ chunkFinal) {
    int tid   = blockIdx.x * blockDim.x + threadIdx.x;   // 65536
    int k     = tid & (NBLK - 1);
    int chunk = (tid >> 7) & (NCH - 1);
    int b     = tid >> 13;
    float c = A[k * 4 + 0];
    float s = A[k * 4 + 1];
    float xr = 0.f, xi = 0.f;
    const float* p = x + ((size_t)(b * SEQ + chunk * CHL) * DIM) + 2 * k;
    for (int t = 0; t < CHL; ++t) {
        float2 u = *(const float2*)p;
        float nr = fmaf(xr, c, fmaf(-xi, s, u.x));
        float ni = fmaf(xr, s, fmaf( xi, c, u.y));
        xr = nr; xi = ni;
        p += DIM;
    }
    ((float2*)chunkFinal)[(b * NCH + chunk) * NBLK + k] = make_float2(xr, xi);
}

// ---------------------------------------------------------------------------
// 2b) Scan over chunk finals: carry[c+1] = carry[c]*R^64 + final[c]; carry[0]=x0
// ---------------------------------------------------------------------------
__global__ void __launch_bounds__(256)
k_scan_chunks(const float* __restrict__ chunkFinal, const float* __restrict__ x0,
              const float* __restrict__ A, float* __restrict__ carry) {
    int tid = blockIdx.x * blockDim.x + threadIdx.x;     // 1024
    int k = tid & (NBLK - 1);
    int b = tid >> 7;
    float cc = A[k * 4 + 0];
    float ss = A[k * 4 + 1];
    #pragma unroll
    for (int i = 0; i < 6; ++i) {                        // R -> R^64 by doubling
        float nc = cc * cc - ss * ss;
        float ns = 2.f * cc * ss;
        cc = nc; ss = ns;
    }
    float xr = x0[(b * NBLK + k) * 2 + 0];
    float xi = x0[(b * NBLK + k) * 2 + 1];
    for (int ch = 0; ch < NCH; ++ch) {
        ((float2*)carry)[(b * NCH + ch) * NBLK + k] = make_float2(xr, xi);
        float2 f = ((const float2*)chunkFinal)[(b * NCH + ch) * NBLK + k];
        float nr = fmaf(xr, cc, fmaf(-xi, ss, f.x));
        float ni = fmaf(xr, ss, fmaf( xi, cc, f.y));
        xr = nr; xi = ni;
    }
}

// ---------------------------------------------------------------------------
// 2c) True recurrence seeded with carry: x[t] = x[t-1]*R + uB[t]
//     writes bf16 x for the C GEMM; chunk 63 also writes fp32 new_state
// ---------------------------------------------------------------------------
__global__ void __launch_bounds__(256)
k_scan_emit(const float* __restrict__ uB, const float* __restrict__ A,
            const float* __restrict__ carry, uint32_t* __restrict__ xbf,
            float* __restrict__ newState) {
    int tid   = blockIdx.x * blockDim.x + threadIdx.x;   // 65536
    int k     = tid & (NBLK - 1);
    int chunk = (tid >> 7) & (NCH - 1);
    int b     = tid >> 13;
    float c = A[k * 4 + 0];
    float s = A[k * 4 + 1];
    float2 ci = ((const float2*)carry)[(b * NCH + chunk) * NBLK + k];
    float xr = ci.x, xi = ci.y;
    size_t row = (size_t)(b * SEQ + chunk * CHL);
    const float* p = uB + row * DIM + 2 * k;
    uint32_t*    q = xbf + row * (DIM / 2) + k;          // one dword = (x_even,x_odd)
    for (int t = 0; t < CHL; ++t) {
        float2 u = *(const float2*)p;
        float nr = fmaf(xr, c, fmaf(-xi, s, u.x));
        float ni = fmaf(xr, s, fmaf( xi, c, u.y));
        *q = pack_bf2(nr, ni);
        xr = nr; xi = ni;
        p += DIM;
        q += DIM / 2;
    }
    if (chunk == NCH - 1)                                 // x[:, -1, :] in fp32
        ((float2*)newState)[b * NBLK + k] = make_float2(xr, xi);
}

// ---------------------------------------------------------------------------
extern "C" void kernel_launch(void* const* d_in, const int* in_sizes, int n_in,
                              void* d_out, int out_size, void* d_ws, size_t ws_size,
                              hipStream_t stream) {
    (void)in_sizes; (void)n_in; (void)out_size; (void)ws_size;
    const float* u  = (const float*)d_in[0];   // 8*4096*256
    const float* x0 = (const float*)d_in[1];   // 8*128*2
    const float* A  = (const float*)d_in[2];   // 128*2*2
    const float* B  = (const float*)d_in[3];   // 256*256
    const float* C  = (const float*)d_in[4];   // 256*256

    float* y        = (float*)d_out;                        // 8*4096*256
    float* newState = y + (size_t)MTOT * DIM;               // 8*256

    char*  ws   = (char*)d_ws;
    float* uBbuf = (float*)ws;                              // 32 MiB (uB, fp32)
    size_t off   = (size_t)MTOT * DIM * sizeof(float);
    uint32_t* xbf = (uint32_t*)(ws + off); off += (size_t)MTOT * (DIM / 2) * 4;  // 16 MiB bf16 x
    uint16_t* Bt = (uint16_t*)(ws + off); off += (size_t)DIM * DIM * 2;
    uint16_t* Ct = (uint16_t*)(ws + off); off += (size_t)DIM * DIM * 2;
    float* chunkFinal = (float*)(ws + off); off += (size_t)BATCH * NCH * NBLK * 2 * 4;
    float* carry      = (float*)(ws + off); off += (size_t)BATCH * NCH * NBLK * 2 * 4;

    // 0) weights -> bf16 transposed
    k_convert_bc<<<256, 256, 0, stream>>>(B, C, Bt, Ct);
    // 1) uB = u @ B   (WMMA bf16, fp32 accumulate; fp32 A converted in-register)
    k_gemm<false><<<dim3(MTOT / 256, DIM / 64), 256, 0, stream>>>(u, Bt, uBbuf);
    // 2) chunked parallel linear scan (rotation recurrence)
    k_scan_local <<<BATCH * NCH * NBLK / 256, 256, 0, stream>>>(uBbuf, A, chunkFinal);
    k_scan_chunks<<<BATCH * NBLK / 256,       256, 0, stream>>>(chunkFinal, x0, A, carry);
    k_scan_emit  <<<BATCH * NCH * NBLK / 256, 256, 0, stream>>>(uBbuf, A, carry, xbf, newState);
    // 3) y = x @ C   (WMMA bf16, bf16 A loaded directly)
    k_gemm<true><<<dim3(MTOT / 256, DIM / 64), 256, 0, stream>>>(xbf, Ct, y);
}
